// RNNAggregator_17111149707398
// MI455X (gfx1250) — matrix operand order repaired and checked by reading, not compiled, not run on hardware
//
#include <hip/hip_runtime.h>

// MI455X / gfx1250: wave32, WMMA 16x16x32 bf16 with f32 accumulate.
typedef __attribute__((ext_vector_type(16))) __bf16 v16bf;
typedef __attribute__((ext_vector_type(8)))  __bf16 v8bf;
typedef __attribute__((ext_vector_type(8)))  float  v8f;

#define N_NODES 16384
#define T_STEPS 25
#define F_DIM   256
#define UNITS   256
#define OUT_DIM 256
#define ROWS_PB 32          // rows per block: 2 row-tiles of 16 per wave
#define BLOCK_THREADS 512   // 16 waves: wave w owns unit-tile w (16 units)

static __device__ inline v8f wmma_bf16(v16bf a, v16bf b, v8f c) {
  // (neg_a, A, neg_b, B, c_mod, C, reuse_a, reuse_b)
  return __builtin_amdgcn_wmma_f32_16x16x32_bf16(false, a, false, b, (short)0, c,
                                                 false, false);
}

// A-fragment (16x32 bf16, MxK) for one 16-row tile, layout used by this file:
//   lane L: m = L%16, half = L/16
//   e=0..15 -> K = kb*32 + half*16 + e (contiguous, matches B fragments)
static __device__ inline v16bf load_a_frag(const __bf16* row, int kb, int half) {
  const __bf16* p = row + kb * 32 + half * 16;
  v8bf lo = *(const v8bf*)(p);
  v8bf hi = *(const v8bf*)(p + 8);
  return __builtin_shufflevector(lo, hi, 0, 1, 2, 3, 4, 5, 6, 7,
                                 8, 9, 10, 11, 12, 13, 14, 15);
}

// Fast gate nonlinearities on gfx1250 transcendental HW (v_exp/v_rcp/v_tanh
// co-execute with WMMA per ISA 8.4 note on TRANS-tracked XDL ops).
#define LOG2E 1.44269504088896340736f
static __device__ inline float fsig(float x) {
  float e = __builtin_amdgcn_exp2f(x * -LOG2E);
  return __builtin_amdgcn_rcpf(1.0f + e);
}
static __device__ inline float ftanh(float x) {
#if __has_builtin(__builtin_amdgcn_tanhf)
  return __builtin_amdgcn_tanhf(x);
#else
  float e = __builtin_amdgcn_exp2f(x * (2.0f * LOG2E));
  return 1.0f - 2.0f * __builtin_amdgcn_rcpf(e + 1.0f);
#endif
}

// ---------------------------------------------------------------------------
// Pre-convert W [256,1024], U [256,1024], Wd [512,256] fp32 -> bf16 fragments.
// B-fragment layout (32x16 bf16, KxN): lane L holds 16 contiguous bf16 with
//   K = kb*32 + (L/16)*16 + e, N = L%16  ->  stored [nt][kb][lane][e].
// ---------------------------------------------------------------------------
#define WU_FRAG_ELEMS (64 * 8 * 32 * 16)   // 262144 per matrix
#define WD_FRAG_ELEMS (16 * 16 * 32 * 16)  // 131072

__global__ void convert_weights_kernel(const float* __restrict__ W,
                                       const float* __restrict__ U,
                                       const float* __restrict__ Wd,
                                       __bf16* __restrict__ fragW,
                                       __bf16* __restrict__ fragU,
                                       __bf16* __restrict__ fragWd) {
  int g = blockIdx.x * blockDim.x + threadIdx.x;
  if (g < 2 * WU_FRAG_ELEMS) {
    int f = g & (WU_FRAG_ELEMS - 1);
    const float* src = (g < WU_FRAG_ELEMS) ? W : U;
    __bf16* dst = (g < WU_FRAG_ELEMS) ? fragW : fragU;
    int e = f & 15, lane = (f >> 4) & 31, kb = (f >> 9) & 7, nt = f >> 12;
    int half = lane >> 4, n = lane & 15;
    int k = kb * 32 + half * 16 + e;
    int col = nt * 16 + n;
    dst[f] = (__bf16)src[k * (4 * UNITS) + col];
  } else {
    int f = g - 2 * WU_FRAG_ELEMS;
    if (f < WD_FRAG_ELEMS) {
      int e = f & 15, lane = (f >> 4) & 31, kb = (f >> 9) & 15, nt = f >> 13;
      int half = lane >> 4, n = lane & 15;
      int k = kb * 32 + half * 16 + e;
      int col = nt * 16 + n;
      fragWd[f] = (__bf16)Wd[k * OUT_DIM + col];
    }
  }
}

// ---------------------------------------------------------------------------
// Persistent LSTM: one block (16 waves) owns 32 rows for all 25 timesteps.
// Wave w owns gate columns {g*256 + w*16 + n} for g=0..3 -> i/f/g/o for a
// (row,unit) live in the same lane -> gate math + c state stay in registers.
// Weight fragments stream from L2 every step. A per-timestep asm-laundered
// SGPR *offset* (not pointer!) blocks LICM from hoisting the t-invariant
// weight loads (which spilled), while keeping the base pointer's provenance
// so the loads stay global_load_b128 (not flat).
// Fragment address = base + g2*131072B + kb*1024B (both fit 24-bit IOFFSET).
// ---------------------------------------------------------------------------
__global__ void __launch_bounds__(BLOCK_THREADS)
lstm_kernel(const float* __restrict__ neigh,
            const __bf16* __restrict__ fragW,
            const __bf16* __restrict__ fragU,
            const float* __restrict__ b,
            __bf16* __restrict__ hT) {
  __shared__ alignas(16) __bf16 xs[ROWS_PB][F_DIM];
  __shared__ alignas(16) __bf16 hs[ROWS_PB][UNITS];

  const int tid  = threadIdx.x;
  const int wave = tid >> 5;           // 0..15 == unit-tile index
  const int lane = tid & 31;
  const int half = lane >> 4;
  const int n    = lane & 15;
  const int rowbase = blockIdx.x * ROWS_PB;

  // Per-wave fragment base offset (elements): wave*4096 + lane*16
  const size_t wbase = (size_t)wave * 4096 + (size_t)lane * 16;

  // h0 = 0
  {
    int row = tid >> 4;                // 0..31
    int c0  = (tid & 15) * 16;
    #pragma unroll
    for (int j = 0; j < 16; ++j) hs[row][c0 + j] = (__bf16)0.0f;
  }

  float creg[2][8];   // [row-tile][r]
  #pragma unroll
  for (int rt = 0; rt < 2; ++rt)
    #pragma unroll
    for (int r = 0; r < 8; ++r) creg[rt][r] = 0.0f;

  #pragma unroll 1
  for (int t = 0; t < T_STEPS; ++t) {
    // Laundered scalar offset: defeats LICM across t without destroying the
    // pointer provenance (keeps global address space inference intact).
    size_t zoff = 0;
    asm volatile("" : "+s"(zoff));
    const __bf16* pw = fragW + zoff + wbase;  // [g2*65536 + kb*512] elements
    const __bf16* pu = fragU + zoff + wbase;

    // stage x_t (fp32 -> bf16) into LDS; prefetch next timestep slab
    {
      int row = tid >> 4;
      int c0  = (tid & 15) * 16;
      const float* src =
          neigh + ((size_t)(rowbase + row) * T_STEPS + t) * F_DIM + c0;
      #pragma unroll
      for (int j = 0; j < 16; ++j) xs[row][c0 + j] = (__bf16)src[j];
      if (t + 1 < T_STEPS) __builtin_prefetch(src + F_DIM, 0, 0);
    }
    __syncthreads();

    // acc = bias broadcast; acc[rt][gate]
    v8f acc[2][4];
    #pragma unroll
    for (int g2 = 0; g2 < 4; ++g2) {
      float bv = b[g2 * UNITS + wave * 16 + n];
      v8f av = {bv, bv, bv, bv, bv, bv, bv, bv};
      acc[0][g2] = av;
      acc[1][g2] = av;
    }

    // z += x_t @ W   (unroll 2: bounds in-flight B-fragment registers)
    #pragma unroll 2
    for (int kb = 0; kb < 8; ++kb) {
      v16bf a0 = load_a_frag(&xs[n][0], kb, half);
      v16bf a1 = load_a_frag(&xs[16 + n][0], kb, half);
      #pragma unroll
      for (int g2 = 0; g2 < 4; ++g2) {
        v16bf bf = *(const v16bf*)(pw + g2 * 65536 + kb * 512);
        acc[0][g2] = wmma_bf16(a0, bf, acc[0][g2]);
        acc[1][g2] = wmma_bf16(a1, bf, acc[1][g2]);
      }
    }
    // z += h_{t-1} @ U
    #pragma unroll 2
    for (int kb = 0; kb < 8; ++kb) {
      v16bf a0 = load_a_frag(&hs[n][0], kb, half);
      v16bf a1 = load_a_frag(&hs[16 + n][0], kb, half);
      #pragma unroll
      for (int g2 = 0; g2 < 4; ++g2) {
        v16bf bf = *(const v16bf*)(pu + g2 * 65536 + kb * 512);
        acc[0][g2] = wmma_bf16(a0, bf, acc[0][g2]);
        acc[1][g2] = wmma_bf16(a1, bf, acc[1][g2]);
      }
    }
    __syncthreads();  // all reads of hs complete before overwrite

    // gates (Keras order i,f,g,o), c/h update; write h bf16 to LDS
    #pragma unroll
    for (int rt = 0; rt < 2; ++rt)
      #pragma unroll
      for (int r = 0; r < 8; ++r) {
        float iv = fsig(acc[rt][0][r]);
        float fv = fsig(acc[rt][1][r]);
        float gv = ftanh(acc[rt][2][r]);
        float ov = fsig(acc[rt][3][r]);
        float cn = fv * creg[rt][r] + iv * gv;
        creg[rt][r] = cn;
        float hv = ov * ftanh(cn);
        hs[rt * 16 + half * 8 + r][wave * 16 + n] = (__bf16)hv;
      }
    __syncthreads();
  }

  // write h_T (bf16) to workspace for the dense kernel
  {
    int row = tid >> 4;
    int c0  = (tid & 15) * 16;
    #pragma unroll
    for (int j = 0; j < 16; ++j)
      hT[(size_t)(rowbase + row) * UNITS + c0 + j] = hs[row][c0 + j];
  }
}

// ---------------------------------------------------------------------------
// out = relu(concat(self, hT) @ Wd + bd)   (relu(relu(x)) == relu(x))
// ---------------------------------------------------------------------------
__global__ void __launch_bounds__(BLOCK_THREADS)
dense_kernel(const float* __restrict__ selfn,
             const __bf16* __restrict__ hT,
             const __bf16* __restrict__ fragWd,
             const float* __restrict__ bd,
             float* __restrict__ out) {
  __shared__ alignas(16) __bf16 cs[ROWS_PB][F_DIM + UNITS];

  const int tid  = threadIdx.x;
  const int wave = tid >> 5;           // 0..15 == output col-tile index
  const int lane = tid & 31;
  const int half = lane >> 4;
  const int n    = lane & 15;
  const int rowbase = blockIdx.x * ROWS_PB;

  // stage concat row: [0,256) = self (fp32->bf16), [256,512) = hT (bf16)
  {
    int row = tid >> 4;
    int c0  = (tid & 15) * 32;
    if (c0 < F_DIM) {
      const float* src = selfn + (size_t)(rowbase + row) * F_DIM + c0;
      #pragma unroll
      for (int j = 0; j < 32; ++j) cs[row][c0 + j] = (__bf16)src[j];
    } else {
      const __bf16* src = hT + (size_t)(rowbase + row) * UNITS + (c0 - F_DIM);
      #pragma unroll
      for (int j = 0; j < 32; ++j) cs[row][c0 + j] = src[j];
    }
  }
  __syncthreads();

  v8f acc[2];  // [row-tile]
  {
    float bv = bd[wave * 16 + n];
    v8f av = {bv, bv, bv, bv, bv, bv, bv, bv};
    acc[0] = av;
    acc[1] = av;
  }

  // base + kb*512 elements (1KB steps, constant IOFFSETs after unroll)
  const __bf16* pd = fragWd + (size_t)wave * 8192 + (size_t)lane * 16;
  #pragma unroll 4
  for (int kb = 0; kb < 16; ++kb) {  // K = 512
    v16bf a0 = load_a_frag(&cs[n][0], kb, half);
    v16bf a1 = load_a_frag(&cs[16 + n][0], kb, half);
    v16bf bf = *(const v16bf*)(pd + kb * 512);
    acc[0] = wmma_bf16(a0, bf, acc[0]);
    acc[1] = wmma_bf16(a1, bf, acc[1]);
  }

  #pragma unroll
  for (int rt = 0; rt < 2; ++rt)
    #pragma unroll
    for (int r = 0; r < 8; ++r) {
      float v = acc[rt][r];
      v = v > 0.0f ? v : 0.0f;
      out[(size_t)(rowbase + rt * 16 + half * 8 + r) * OUT_DIM + wave * 16 + n] =
          v;
    }
}

extern "C" void kernel_launch(void* const* d_in, const int* in_sizes, int n_in,
                              void* d_out, int out_size, void* d_ws,
                              size_t ws_size, hipStream_t stream) {
  const float* selfn = (const float*)d_in[0];
  const float* neigh = (const float*)d_in[1];
  // d_in[2] (len_adj_nodes) unused by the reference body.
  const float* W  = (const float*)d_in[3];
  const float* U  = (const float*)d_in[4];
  const float* b  = (const float*)d_in[5];
  const float* Wd = (const float*)d_in[6];
  const float* bd = (const float*)d_in[7];

  // ws layout: [fragW 512K][fragU 512K][fragWd 256K][hT bf16 8M] = 9.25 MB
  char* ws = (char*)d_ws;
  __bf16* fragW  = (__bf16*)(ws);
  __bf16* fragU  = (__bf16*)(ws + (size_t)512 * 1024);
  __bf16* fragWd = (__bf16*)(ws + (size_t)1024 * 1024);
  __bf16* hT     = (__bf16*)(ws + (size_t)1024 * 1024 + (size_t)256 * 1024);

  // 2*262144 + 131072 = 655360 fragment elements -> 2560 blocks of 256
  convert_weights_kernel<<<2560, 256, 0, stream>>>(W, U, Wd, fragW, fragU,
                                                   fragWd);
  lstm_kernel<<<N_NODES / ROWS_PB, BLOCK_THREADS, 0, stream>>>(neigh, fragW,
                                                               fragU, b, hT);
  dense_kernel<<<N_NODES / ROWS_PB, BLOCK_THREADS, 0, stream>>>(
      selfn, hT, fragWd, bd, (float*)d_out);
}
